// FusedDecompressDistance_43748536877554
// MI455X (gfx1250) — matrix operand order repaired and checked by reading, not compiled, not run on hardware
//
#include <hip/hip_runtime.h>
#include <hip/hip_bf16.h>

#define EPSF 1e-6f

typedef __attribute__((ext_vector_type(16))) __bf16 v16bf;
typedef __attribute__((ext_vector_type(8)))  __bf16 v8bf;
typedef __attribute__((ext_vector_type(8)))  float  v8f;

union BF16Frag { v16bf v; v8bf h[2]; };

// Async global->LDS staging via inline asm (GLOBAL_LOAD_ASYNC_TO_LDS_B128,
// tracked by ASYNCcnt).  Set to 0 to fall back to VGPR-bounce staging.
#define USE_ASYNC_LDS 1

// Stage 32 bytes (16 bf16) from global to LDS, per lane.
__device__ __forceinline__ void stage32B(__bf16* lds_dst, const __bf16* gsrc) {
#if USE_ASYNC_LDS
  // Low 32 bits of the flat shared-aperture address are the LDS byte offset.
  unsigned lds_addr = (unsigned)(size_t)(void*)lds_dst;
  asm volatile(
      "global_load_async_to_lds_b128 %0, %1, off\n\t"
      "global_load_async_to_lds_b128 %0, %1, off offset:16"
      :
      : "v"(lds_addr), "v"(gsrc)
      : "memory");
#else
  const uint4* g = (const uint4*)gsrc;
  ((uint4*)lds_dst)[0] = g[0];
  ((uint4*)lds_dst)[1] = g[1];
#endif
}

__device__ __forceinline__ void async_wait_all() {
#if USE_ASYNC_LDS
  asm volatile("s_wait_asynccnt 0x0" ::: "memory");
#endif
}

// ---------------------------------------------------------------------------
// Elementwise prep kernels
// ---------------------------------------------------------------------------
__global__ __launch_bounds__(256) void f32_to_bf16_kernel(const float* __restrict__ in,
                                                          __bf16* __restrict__ out,
                                                          size_t n) {
  size_t i = (size_t)blockIdx.x * blockDim.x + threadIdx.x;
  if (i < n) out[i] = (__bf16)in[i];
}

__global__ __launch_bounds__(256) void dequant_kernel(const int* __restrict__ codes,
                                                      const float* __restrict__ scale,
                                                      const float* __restrict__ zero,
                                                      __bf16* __restrict__ out,
                                                      int Dc, size_t n) {
  size_t i = (size_t)blockIdx.x * blockDim.x + threadIdx.x;
  if (i < n) {
    size_t row = i / (size_t)Dc;  // (b*Lk + lk)
    float v = scale[row] * ((float)codes[i] - zero[row]);
    out[i] = (__bf16)v;
  }
}

// One wave per row, shuffle reduction.
__global__ __launch_bounds__(256) void row_sumsq_f32_kernel(const float* __restrict__ X,
                                                            float* __restrict__ out,
                                                            int ncols) {
  int wave = threadIdx.x >> 5, lane = threadIdx.x & 31;
  size_t row = (size_t)blockIdx.x * 8 + wave;
  const float* p = X + row * (size_t)ncols;
  float s = 0.f;
  for (int i = lane; i < ncols; i += 32) { float v = p[i]; s += v * v; }
  for (int m = 16; m >= 1; m >>= 1) s += __shfl_xor(s, m, 32);
  if (lane == 0) out[row] = s;
}

__global__ __launch_bounds__(256) void row_sumsq_bf16_kernel(const __bf16* __restrict__ X,
                                                             float* __restrict__ out,
                                                             int ncols) {
  int wave = threadIdx.x >> 5, lane = threadIdx.x & 31;
  size_t row = (size_t)blockIdx.x * 8 + wave;
  const __bf16* p = X + row * (size_t)ncols;
  float s = 0.f;
  for (int i = lane; i < ncols; i += 32) { float v = (float)p[i]; s += v * v; }
  for (int m = 16; m >= 1; m >>= 1) s += __shfl_xor(s, m, 32);
  if (lane == 0) out[row] = s;
}

// ---------------------------------------------------------------------------
// Shared WMMA GEMM core:  C(128x128) = A(128xK) * B(128xK)^T per block,
// bf16 in, f32 acc.  8 waves, each owning a 64x32 region (4x2 tiles 16x16).
// BK = 32 (one v_wmma_f32_16x16x32_bf16 K-step), double-buffered LDS with
// async global->LDS staging overlapping the WMMAs of the previous tile.
// ---------------------------------------------------------------------------
__device__ __forceinline__ void gemm_core_nt(const __bf16* __restrict__ A, int lda,
                                             const __bf16* __restrict__ Bm, int ldb,
                                             int K,
                                             __bf16 (*As)[128][32],
                                             __bf16 (*Bs)[128][32],
                                             v8f acc[4][2]) {
  const int tid  = threadIdx.x;
  const int lane = tid & 31;
  const int wave = tid >> 5;
  const int wm   = (wave & 1) * 64;   // wave M offset in tile
  const int wn   = (wave >> 1) * 32;  // wave N offset in tile
  const int lr   = tid >> 1;          // staging row 0..127
  const int lc   = (tid & 1) * 16;    // staging col (halves) 0 or 16
  const int l16  = lane & 15;
  const int hh   = lane >> 4;

  v8f zero = {};
  for (int mi = 0; mi < 4; ++mi)
    for (int ni = 0; ni < 2; ++ni) acc[mi][ni] = zero;

  const __bf16* ga = A + (size_t)lr * lda + lc;
  const __bf16* gb = Bm + (size_t)lr * ldb + lc;

  const int nk = K >> 5;
  // Prefetch tile 0 into buffer 0.
  stage32B(&As[0][lr][lc], ga);
  stage32B(&Bs[0][lr][lc], gb);

  for (int i = 0; i < nk; ++i) {
    async_wait_all();       // all of this wave's async copies landed in LDS
    __syncthreads();        // all waves' copies landed; prev reads complete

    if (i + 1 < nk) {       // overlap next tile's DMA with this tile's WMMAs
      const int kt = (i + 1) << 5;
      const int nb = (i + 1) & 1;
      stage32B(&As[nb][lr][lc], ga + kt);
      stage32B(&Bs[nb][lr][lc], gb + kt);
    }

    __bf16 (*Asc)[32] = As[i & 1];
    __bf16 (*Bsc)[32] = Bs[i & 1];

    // Fragments per ISA 7.12.2 bf16 layouts (ds_load_b128 pairs).
    BF16Frag af[4], bf[2];
    for (int mi = 0; mi < 4; ++mi) {
      const int m = wm + mi * 16 + l16;
      af[mi].h[0] = *(const v8bf*)&Asc[m][hh * 8];
      af[mi].h[1] = *(const v8bf*)&Asc[m][16 + hh * 8];
    }
    for (int ni = 0; ni < 2; ++ni) {
      const int n = wn + ni * 16 + l16;
      bf[ni].h[0] = *(const v8bf*)&Bsc[n][hh * 8];
      bf[ni].h[1] = *(const v8bf*)&Bsc[n][16 + hh * 8];
    }

    for (int mi = 0; mi < 4; ++mi)
      for (int ni = 0; ni < 2; ++ni)
        acc[mi][ni] = __builtin_amdgcn_wmma_f32_16x16x32_bf16(
            false, af[mi].v, false, bf[ni].v, (short)0, acc[mi][ni],
            false, false);
  }
}

// ---------------------------------------------------------------------------
// Kernel: decompress K.  k = kc @ W_up^T.  A = kc (Lk x Dc), B = W_up (D x Dc).
// ---------------------------------------------------------------------------
__global__ __launch_bounds__(256) void gemm_decompress_kernel(
    const __bf16* __restrict__ kc, const __bf16* __restrict__ W,
    __bf16* __restrict__ Kout, int Lk, int D, int Dc) {
  __shared__ __align__(16) __bf16 As[2][128][32];
  __shared__ __align__(16) __bf16 Bs[2][128][32];
  const int b    = blockIdx.z;
  const int row0 = blockIdx.y * 128;  // along Lk
  const int col0 = blockIdx.x * 128;  // along D
  v8f acc[4][2];
  gemm_core_nt(kc + ((size_t)b * Lk + row0) * Dc, Dc,
               W + (size_t)col0 * Dc, Dc, Dc, As, Bs, acc);

  const int lane = threadIdx.x & 31;
  const int wave = threadIdx.x >> 5;
  const int wm = (wave & 1) * 64, wn = (wave >> 1) * 32;
  const int l16 = lane & 15, hh = lane >> 4;
  for (int mi = 0; mi < 4; ++mi)
    for (int ni = 0; ni < 2; ++ni)
      for (int j = 0; j < 8; ++j) {
        int m = row0 + wm + mi * 16 + j + 8 * hh;
        int n = col0 + wn + ni * 16 + l16;
        Kout[((size_t)b * Lk + m) * (size_t)D + n] = (__bf16)acc[mi][ni][j];
      }
}

// ---------------------------------------------------------------------------
// Kernel: cross GEMM + fused Poincare distance epilogue.
// cross = q @ k^T ; dist = acosh(1 + 2*max(qs+ks-2c,0)/((1-qcl)(1-kcl)+eps)).
// ---------------------------------------------------------------------------
__global__ __launch_bounds__(256) void cross_dist_kernel(
    const __bf16* __restrict__ Qb, const __bf16* __restrict__ Kb,
    const float* __restrict__ q_sq, const float* __restrict__ k_sq,
    float* __restrict__ out, int Lq, int Lk, int D) {
  __shared__ __align__(16) __bf16 As[2][128][32];
  __shared__ __align__(16) __bf16 Bs[2][128][32];
  const int b    = blockIdx.z;
  const int row0 = blockIdx.y * 128;  // along Lq
  const int col0 = blockIdx.x * 128;  // along Lk
  v8f acc[4][2];
  gemm_core_nt(Qb + ((size_t)b * Lq + row0) * D, D,
               Kb + ((size_t)b * Lk + col0) * D, D, D, As, Bs, acc);

  const int lane = threadIdx.x & 31;
  const int wave = threadIdx.x >> 5;
  const int wm = (wave & 1) * 64, wn = (wave >> 1) * 32;
  const int l16 = lane & 15, hh = lane >> 4;

  float ksv[2], kcl[2];
  for (int ni = 0; ni < 2; ++ni) {
    ksv[ni] = k_sq[(size_t)b * Lk + col0 + wn + ni * 16 + l16];
    kcl[ni] = 1.0f - fminf(ksv[ni], 1.0f - EPSF);
  }
  for (int mi = 0; mi < 4; ++mi) {
    float qsv[8];
    for (int j = 0; j < 8; ++j)
      qsv[j] = q_sq[(size_t)b * Lq + row0 + wm + mi * 16 + j + 8 * hh];
    for (int ni = 0; ni < 2; ++ni) {
      for (int j = 0; j < 8; ++j) {
        float qs = qsv[j];
        float cr = acc[mi][ni][j];
        float diff = fmaxf(qs + ksv[ni] - 2.0f * cr, 0.0f);
        float denom = (1.0f - fminf(qs, 1.0f - EPSF)) * kcl[ni] + EPSF;
        float x = 1.0f + 2.0f * diff / denom;
        int m = row0 + wm + mi * 16 + j + 8 * hh;
        int n = col0 + wn + ni * 16 + l16;
        out[((size_t)b * Lq + m) * (size_t)Lk + n] = acoshf(x);
      }
    }
  }
}

// ---------------------------------------------------------------------------
// Host-side launch
// ---------------------------------------------------------------------------
extern "C" void kernel_launch(void* const* d_in, const int* in_sizes, int n_in,
                              void* d_out, int out_size, void* d_ws, size_t ws_size,
                              hipStream_t stream) {
  const float* q      = (const float*)d_in[0];   // (B,Lq,D) f32
  const int*   kcode  = (const int*)d_in[1];     // (B,Lk,Dc) i32
  const float* W      = (const float*)d_in[2];   // (D,Dc) f32
  const float* kscale = (const float*)d_in[3];   // (B,Lk,1) f32
  const float* kzero  = (const float*)d_in[4];   // (B,Lk,1) f32

  const int B = 4, Lq = 2048, Lk = 4096, D = 1024, Dc = 512;

  char* ws = (char*)d_ws;
  size_t off = 0;
  auto alloc = [&](size_t bytes) {
    void* p = ws + off;
    off = (off + bytes + 255) & ~(size_t)255;
    return p;
  };
  __bf16* qb  = (__bf16*)alloc((size_t)B * Lq * D * 2);   // 16 MB
  __bf16* kb  = (__bf16*)alloc((size_t)B * Lk * D * 2);   // 32 MB
  __bf16* kcq = (__bf16*)alloc((size_t)B * Lk * Dc * 2);  // 16 MB
  __bf16* wb  = (__bf16*)alloc((size_t)D * Dc * 2);       // 1 MB
  float*  qsq = (float*)alloc((size_t)B * Lq * 4);
  float*  ksq = (float*)alloc((size_t)B * Lk * 4);

  size_t nQ = (size_t)B * Lq * D;
  size_t nW = (size_t)D * Dc;
  size_t nKC = (size_t)B * Lk * Dc;

  f32_to_bf16_kernel<<<(unsigned)((nQ + 255) / 256), 256, 0, stream>>>(q, qb, nQ);
  f32_to_bf16_kernel<<<(unsigned)((nW + 255) / 256), 256, 0, stream>>>(W, wb, nW);
  dequant_kernel<<<(unsigned)((nKC + 255) / 256), 256, 0, stream>>>(kcode, kscale, kzero, kcq, Dc, nKC);
  row_sumsq_f32_kernel<<<(B * Lq) / 8, 256, 0, stream>>>(q, qsq, D);

  gemm_decompress_kernel<<<dim3(D / 128, Lk / 128, B), 256, 0, stream>>>(kcq, wb, kb, Lk, D, Dc);
  row_sumsq_bf16_kernel<<<(B * Lk) / 8, 256, 0, stream>>>(kb, ksq, D);

  cross_dist_kernel<<<dim3(Lk / 128, Lq / 128, B), 256, 0, stream>>>(
      qb, kb, qsq, ksq, (float*)d_out, Lq, Lk, D);
}
